// CRKT_24661702214063
// MI455X (gfx1250) — matrix operand order repaired and checked by reading, not compiled, not run on hardware
//
#include <hip/hip_runtime.h>
#include <hip/hip_bf16.h>

// ---------------------------------------------------------------------------
// CRKT forward for MI455X (gfx1250, wave32, WMMA bf16 w/ fp32 accumulate)
// WMMA operands are bf16; global ones are packed fragment loads, attn probs are converted from f32 LDS.
// ---------------------------------------------------------------------------

static constexpr int B_  = 32;
static constexpr int S_  = 512;
static constexpr int D_  = 256;
static constexpr int H_  = 8;
static constexpr int DK_ = 32;
static constexpr int NO_ = 4;
static constexpr int NQ_ = 2000;
static constexpr int BS_ = B_ * S_;               // 16384 rows
static constexpr size_t BSD_ = (size_t)BS_ * D_;  // 4,194,304
static constexpr float LAMB_ = 0.3f;
static constexpr float TEMP_ = 0.05f;

typedef __attribute__((ext_vector_type(16))) __bf16 v16bf;
typedef __attribute__((ext_vector_type(8)))  __bf16 v8bf;
typedef __attribute__((ext_vector_type(8)))  float  v8f;

// GEMM output modes
static constexpr int OM_F32 = 0;       // f32 row-major
static constexpr int OM_BF16 = 1;      // bf16 row-major
static constexpr int OM_VT = 2;        // bf16 head-transposed [b,h,dk,s]
static constexpr int OM_RELU_BF16 = 3; // relu + bf16 row-major

__device__ __forceinline__ v8f wmma_bf16(v16bf a, v16bf b, v8f c) {
  return __builtin_amdgcn_wmma_f32_16x16x32_bf16(
      false, a, false, b, (short)0, c, false, false);
}

__device__ __forceinline__ v16bf join16(v8bf a0, v8bf a1) {
  v16bf r;
#pragma unroll
  for (int e = 0; e < 8; ++e) { r[e] = a0[e]; r[e + 8] = a1[e]; }
  return r;
}

// ---------------------------------------------------------------------------
// Weight pre-swizzle: W f32 [K,N] -> bf16 [K/32][N][32] (contiguous fragments)
// ---------------------------------------------------------------------------
__global__ __launch_bounds__(256)
void swz_kernel(const float* __restrict__ W, __bf16* __restrict__ out, int N) {
  size_t id = (size_t)blockIdx.x * 256 + threadIdx.x;   // over K*N
  int k = (int)(id / N), n = (int)(id % N);
  out[(size_t)(k >> 5) * N * 32 + (size_t)n * 32 + (k & 31)] = (__bf16)W[id];
}

__global__ __launch_bounds__(256)
void conv_bf_kernel(const float* __restrict__ in, __bf16* __restrict__ out) {
  size_t i = (size_t)blockIdx.x * 256 + threadIdx.x;
  out[i] = (__bf16)in[i];
}

// ---------------------------------------------------------------------------
// GEMM: C[M,N] = act(A[M,K] @ W[K,N] + bias[N]); K%32==0, M%16, N%64.
// A: bf16 row-major. Wz: bf16 swizzled [K/32][N][32].
// One wave computes a 16x64 strip; all fragments are packed b128 loads.
// ---------------------------------------------------------------------------
__global__ __launch_bounds__(32)
void gemm_kernel(const __bf16* __restrict__ A, const __bf16* __restrict__ Wz,
                 const float* __restrict__ bias, void* __restrict__ Cout,
                 int M, int N, int K, int mode) {
  const int tm  = blockIdx.x;
  const int tn0 = blockIdx.y * 4;    // four consecutive 16-col tiles
  const int lane = threadIdx.x;
  const int row  = lane & 15;
  const int hi   = lane >> 4;        // 0: lanes 0-15, 1: lanes 16-31
  v8f acc0 = {}, acc1 = {}, acc2 = {}, acc3 = {};
  const __bf16* arow = A + (size_t)(tm * 16 + row) * K + hi * 16;
  const __bf16* wz   = Wz + (size_t)(tn0 * 16 + row) * 32 + hi * 16;
  const size_t wstep = (size_t)N * 32;
  for (int k0 = 0; k0 < K; k0 += 32) {
    // A 16x32 fragment: two contiguous 8-half groups
    v16bf af = join16(*(const v8bf*)(arow + k0), *(const v8bf*)(arow + k0 + 8));
    const __bf16* wk = wz + (size_t)(k0 >> 5) * wstep;
    acc0 = wmma_bf16(af, *(const v16bf*)(wk),           acc0);
    acc1 = wmma_bf16(af, *(const v16bf*)(wk + 16 * 32), acc1);
    acc2 = wmma_bf16(af, *(const v16bf*)(wk + 32 * 32), acc2);
    acc3 = wmma_bf16(af, *(const v16bf*)(wk + 48 * 32), acc3);
  }
#pragma unroll
  for (int t = 0; t < 4; ++t) {
    v8f acc = (t == 0) ? acc0 : (t == 1) ? acc1 : (t == 2) ? acc2 : acc3;
    const int ncol = tn0 * 16 + t * 16 + row;
    const float bv = bias[ncol];
#pragma unroll
    for (int r = 0; r < 8; ++r) {
      const int m = tm * 16 + r + (hi ? 8 : 0);
      float v = acc[r] + bv;
      if (mode == OM_RELU_BF16) v = fmaxf(v, 0.f);
      if (mode == OM_F32) {
        ((float*)Cout)[(size_t)m * N + ncol] = v;
      } else if (mode == OM_VT) {
        // m = b*S + s ; ncol = h*32 + dk  ->  [b,h,dk,s]
        const int bb = m >> 9, s = m & (S_ - 1), hh = ncol >> 5, dk = ncol & 31;
        ((__bf16*)Cout)[((size_t)(bb * H_ + hh) * DK_ + dk) * S_ + s] = (__bf16)v;
      } else {
        ((__bf16*)Cout)[(size_t)m * N + ncol] = (__bf16)v;
      }
    }
  }
}

// ---------------------------------------------------------------------------
// Fused KT attention core. One workgroup (256 thr = 8 waves) per (b,h,qtile16).
// qk: bf16 row-major [b,s,h*32+d] (Q and K fragments are packed loads).
// vT: bf16 [b,h,dk,s]. Output: bf16 row-major [b,s,h*32+d].
// ---------------------------------------------------------------------------
__global__ __launch_bounds__(256)
void attn_kernel(const __bf16* __restrict__ qk, const __bf16* __restrict__ vT,
                 __bf16* __restrict__ ob, const float* __restrict__ decay_p,
                 int maxout) {
  __shared__ float sc[16][S_ + 1];
  __shared__ float pbuf[8][16][16];
  const int qt = blockIdx.x;          // S/16 tiles
  const int h  = blockIdx.y;
  const int b  = blockIdx.z;
  const int tid  = threadIdx.x;
  const int wave = tid >> 5;
  const int lane = tid & 31;
  const int hi   = lane >> 4;
  const int lrow = lane & 15;
  const int i0 = qt * 16;
  const float rs = 0.1767766952966369f; // 1/sqrt(DK)

  // ---- Phase 1: scores = Q(16x32) @ K^T ----
  {
    const __bf16* qrow = qk + (size_t)(b * S_ + i0 + lrow) * D_ + h * DK_ + hi * 16;
    v16bf af = join16(*(const v8bf*)qrow, *(const v8bf*)(qrow + 8));
    for (int j = wave; j < S_ / 16; j += 8) {
      const __bf16* krow = qk + (size_t)(b * S_ + j * 16 + lrow) * D_ + h * DK_ + hi * 16;
      v16bf bf = *(const v16bf*)krow;
      v8f c = {};
      c = wmma_bf16(af, bf, c);
#pragma unroll
      for (int r = 0; r < 8; ++r) {
        int m = r + (hi ? 8 : 0);
        sc[m][j * 16 + lrow] = c[r] * rs;
      }
    }
  }
  __syncthreads();

  // ---- Phase 2: row-wise (16 threads per row; each owns 32 contiguous cols)
  {
    const int r = tid >> 4;
    const int g = tid & 15;
    const int iabs = i0 + r;
    const float dec = fabsf(decay_p[h]);
    float s[32], w[32];
#pragma unroll
    for (int c = 0; c < 32; ++c) s[c] = sc[r][g * 32 + c];
    // softmax #1 over j<=iabs
    float mx = -3.4e38f;
#pragma unroll
    for (int c = 0; c < 32; ++c) { int j = g * 32 + c; if (j <= iabs) mx = fmaxf(mx, s[c]); }
    for (int d2 = 8; d2 >= 1; d2 >>= 1) mx = fmaxf(mx, __shfl_xor(mx, d2, 16));
    float sum = 0.f;
#pragma unroll
    for (int c = 0; c < 32; ++c) {
      int j = g * 32 + c;
      float e = (j <= iabs) ? expf(s[c] - mx) : 0.f;
      w[c] = e; sum += e;
    }
    float tot = sum;
    for (int d2 = 8; d2 >= 1; d2 >>= 1) tot += __shfl_xor(tot, d2, 16);
    float inv = 1.f / tot;
    // inclusive cumsum of masked probs across the row
    float run = 0.f;
#pragma unroll
    for (int c = 0; c < 32; ++c) { w[c] *= inv; run += w[c]; w[c] = run; }
    float ts = run, scan = run;
    for (int d2 = 1; d2 < 16; d2 <<= 1) {
      float t = __shfl_up(scan, d2, 16);
      if (g >= d2) scan += t;
    }
    float excl = scan - ts;
    float tsum = __shfl(scan, 15, 16);   // total mass (≈1)
    // decayed scores
#pragma unroll
    for (int c = 0; c < 32; ++c) {
      int j = g * 32 + c;
      float tau  = w[c] + excl;
      float dist = fmaxf(fabsf((float)(j - iabs)) * (tsum - tau), 0.f);
      w[c] = s[c] * expf(-dec * dist);
    }
    // softmax #2
    mx = -3.4e38f;
#pragma unroll
    for (int c = 0; c < 32; ++c) { int j = g * 32 + c; if (j <= iabs) mx = fmaxf(mx, w[c]); }
    for (int d2 = 8; d2 >= 1; d2 >>= 1) mx = fmaxf(mx, __shfl_xor(mx, d2, 16));
    sum = 0.f;
#pragma unroll
    for (int c = 0; c < 32; ++c) {
      int j = g * 32 + c;
      float e = (j <= iabs) ? expf(w[c] - mx) : 0.f;
      w[c] = e; sum += e;
    }
    tot = sum;
    for (int d2 = 8; d2 >= 1; d2 >>= 1) tot += __shfl_xor(tot, d2, 16);
    inv = 1.f / tot;
    float amax = 0.f;
#pragma unroll
    for (int c = 0; c < 32; ++c) { w[c] *= inv; amax = fmaxf(amax, w[c]); }
    if (maxout) {
      for (int d2 = 8; d2 >= 1; d2 >>= 1) amax = fmaxf(amax, __shfl_xor(amax, d2, 16));
      float f = fminf(1.f / amax, 5.f);
#pragma unroll
      for (int c = 0; c < 32; ++c) w[c] *= f;
    }
#pragma unroll
    for (int c = 0; c < 32; ++c) sc[r][g * 32 + c] = w[c];
  }
  __syncthreads();

  // ---- Phase 3: out(16x32) = alpha(16x512) @ V(512x32), all 8 waves ----
  {
    const int t  = wave & 1;        // n-tile (0..1)
    const int kq = wave >> 1;       // k-quarter (0..3)
    const int n0 = t * 16;
    const __bf16* vrow = vT + ((size_t)(b * H_ + h) * DK_ + n0 + lrow) * S_ + hi * 16;
    v8f c = {};
    for (int ks = kq * 4; ks < kq * 4 + 4; ++ks) {
      v16bf af, bf;
#pragma unroll
      for (int e = 0; e < 16; ++e) {
        int kc = e + (hi ? 16 : 0);
        af[e] = (__bf16)sc[lrow][ks * 32 + kc];
      }
      bf = *(const v16bf*)(vrow + ks * 32);
      c = wmma_bf16(af, bf, c);
    }
#pragma unroll
    for (int r = 0; r < 8; ++r) {
      int m = r + (hi ? 8 : 0);
      pbuf[wave][m][lrow] = c[r];
    }
  }
  __syncthreads();
  for (int l = tid; l < 512; l += 256) {
    const int t = l >> 8, m = (l >> 4) & 15, n = l & 15;
    float v = pbuf[t][m][n] + pbuf[t + 2][m][n] + pbuf[t + 4][m][n] + pbuf[t + 6][m][n];
    ob[(size_t)(b * S_ + i0 + m) * D_ + h * DK_ + t * 16 + n] = (__bf16)v;
  }
}

// ---------------------------------------------------------------------------
// LayerNorm in place over D=256 (one block per row)
// ---------------------------------------------------------------------------
__global__ __launch_bounds__(256)
void ln_kernel(float* __restrict__ x, const float* __restrict__ g,
               const float* __restrict__ bta) {
  __shared__ float red[256];
  const size_t row = blockIdx.x;
  const int d = threadIdx.x;
  float v = x[row * D_ + d];
  red[d] = v; __syncthreads();
  for (int s = 128; s > 0; s >>= 1) { if (d < s) red[d] += red[d + s]; __syncthreads(); }
  float mu = red[0] / (float)D_; __syncthreads();
  float c = v - mu;
  red[d] = c * c; __syncthreads();
  for (int s = 128; s > 0; s >>= 1) { if (d < s) red[d] += red[d + s]; __syncthreads(); }
  float var = red[0] / (float)D_;
  x[row * D_ + d] = c * rsqrtf(var + 1e-5f) * g[d] + bta[d];
}

// ---------------------------------------------------------------------------
// Elementwise / gather kernels
// ---------------------------------------------------------------------------
__global__ __launch_bounds__(256)
void gather_q_kernel(const int* __restrict__ question,
                     const float* __restrict__ qemb, __bf16* __restrict__ out) {
  size_t idx = (size_t)blockIdx.x * 256 + threadIdx.x;
  int bs = (int)(idx >> 8), d = (int)(idx & 255);
  int q = question[bs]; int qe = (q >= 0) ? q : NQ_;
  out[idx] = (__bf16)qemb[(size_t)qe * D_ + d];
}

__global__ __launch_bounds__(256)
void gather_ot_kernel(const int* __restrict__ question, const int* __restrict__ option,
                      const float* __restrict__ remb, float* __restrict__ out32,
                      __bf16* __restrict__ out16) {
  size_t idx = (size_t)blockIdx.x * 256 + threadIdx.x;
  int bs = (int)(idx >> 8), d = (int)(idx & 255);
  int q = question[bs], o = option[bs];
  int qe = (q >= 0) ? q : NQ_;
  size_t oidx = (o >= 0) ? ((size_t)qe * NO_ + o) : (size_t)NQ_ * NO_;
  float v = remb[oidx * D_ + d];
  out32[idx] = v;
  out16[idx] = (__bf16)v;
}

__device__ __forceinline__ int ut_mask_val(int q, int op, int o) {
  int qe = (q >= 0) ? q : NQ_;
  int opt = (op >= 0) ? op : 0;
  int om = (qe == NQ_ && o >= 2) ? 0 : 1;
  return om - ((opt == o) ? 1 : 0);
}

__global__ __launch_bounds__(256)
void gather_ut_kernel(const int* __restrict__ question, const int* __restrict__ option,
                      const float* __restrict__ remb, __bf16* __restrict__ out, int o) {
  size_t idx = (size_t)blockIdx.x * 256 + threadIdx.x;
  int bs = (int)(idx >> 8), d = (int)(idx & 255);
  int q = question[bs], op = option[bs];
  int qe = (q >= 0) ? q : NQ_;
  int mask = ut_mask_val(q, op, o);
  size_t uidx = (mask > 0) ? ((size_t)qe * NO_ + o) : ((size_t)NQ_ * NO_ + 1);
  out[idx] = (__bf16)remb[uidx * D_ + d];
}

__global__ __launch_bounds__(256)
void accum_ut_kernel(const int* __restrict__ question, const int* __restrict__ option,
                     const float* __restrict__ up, float* __restrict__ acc,
                     int o, int init) {
  size_t idx = (size_t)blockIdx.x * 256 + threadIdx.x;
  int bs = (int)(idx >> 8);
  float m = (float)ut_mask_val(question[bs], option[bs], o);
  float v = up[idx] * m;
  acc[idx] = init ? v : (acc[idx] + v);
}

__global__ __launch_bounds__(256)
void combine_dt_kernel(const int* __restrict__ question, const int* __restrict__ option,
                       const int* __restrict__ score,
                       const float* __restrict__ ot, const float* __restrict__ mc,
                       const float* __restrict__ mw, float* __restrict__ acc_dt) {
  size_t idx = (size_t)blockIdx.x * 256 + threadIdx.x;
  int bs = (int)(idx >> 8);
  int sc = score[bs];
  float sel = (sc == 1) ? mc[idx] : ((sc == 0) ? mw[idx] : ot[idx]);
  int q = question[bs], op = option[bs];
  float cnt = 0.f;
#pragma unroll
  for (int o = 0; o < NO_; ++o) cnt += (float)ut_mask_val(q, op, o);
  acc_dt[idx] = sel - LAMB_ * acc_dt[idx] / cnt;
}

// masked mean-pool over s in [0, S-1) + L2 normalize. One block per b.
__global__ __launch_bounds__(256)
void pool_kernel(const float* __restrict__ h, const int* __restrict__ score,
                 float* __restrict__ out) {
  __shared__ float red[256];
  const int b = blockIdx.x, d = threadIdx.x;
  float s = 0.f, cnt = 0.f;
  for (int t = 0; t < S_ - 1; ++t) {
    float m = (score[b * S_ + t] != -1) ? 1.f : 0.f;
    s += h[((size_t)b * S_ + t) * D_ + d] * m;
    cnt += m;
  }
  float val = s / cnt;
  red[d] = val * val; __syncthreads();
  for (int st = 128; st > 0; st >>= 1) { if (d < st) red[d] += red[d + st]; __syncthreads(); }
  out[(size_t)b * D_ + d] = val / sqrtf(red[0]);
}

__global__ __launch_bounds__(64)
void cos_kernel(const float* __restrict__ a, const float* __restrict__ bp,
                const float* __restrict__ cn, float* __restrict__ out) {
  const int b = blockIdx.x, j = threadIdx.x;
  const float* x = (j < B_) ? (bp + (size_t)j * D_) : (cn + (size_t)(j - B_) * D_);
  float dot = 0.f;
  for (int d = 0; d < D_; ++d) dot += a[(size_t)b * D_ + d] * x[d];
  float r = dot / TEMP_;
  if (j >= B_ && (j - B_) == b) r += 1.f;
  out[(size_t)b * (2 * B_) + j] = r;
}

// ---------------------------------------------------------------------------
// Host orchestration
// ---------------------------------------------------------------------------
struct Prm {
  const float *qemb, *remb;
  const float *ew1[3], *eb1[3], *ew2[3], *eb2[3];     // correct, wrong, unchosen
  const float *wq[3], *bq[3], *wv[3], *bv[3], *wo[3], *bo[3];
  const float *decay[3], *lng[3], *lnb[3];            // ar, aq, as
  const int *question, *score, *option, *pos_score, *pos_option, *neg_score, *neg_option;
};

extern "C" void kernel_launch(void* const* d_in, const int* in_sizes, int n_in,
                              void* d_out, int out_size, void* d_ws, size_t ws_size,
                              hipStream_t stream) {
  (void)in_sizes; (void)out_size; (void)ws_size;
  Prm p;
  if (n_in >= 49) {
    // params leaves as separate entries, dict insertion order, then int arrays
    int i = 0;
    p.qemb = (const float*)d_in[i++];
    p.remb = (const float*)d_in[i++];
    for (int e = 0; e < 3; ++e) {
      p.ew1[e] = (const float*)d_in[i++]; p.eb1[e] = (const float*)d_in[i++];
      p.ew2[e] = (const float*)d_in[i++]; p.eb2[e] = (const float*)d_in[i++];
    }
    for (int a = 0; a < 3; ++a) {
      p.wq[a] = (const float*)d_in[i++]; p.bq[a] = (const float*)d_in[i++];
      p.wv[a] = (const float*)d_in[i++]; p.bv[a] = (const float*)d_in[i++];
      p.wo[a] = (const float*)d_in[i++]; p.bo[a] = (const float*)d_in[i++];
      p.decay[a] = (const float*)d_in[i++];
      p.lng[a] = (const float*)d_in[i++]; p.lnb[a] = (const float*)d_in[i++];
    }
    p.question  = (const int*)d_in[i++];
    /*concept*/ i++;
    p.score      = (const int*)d_in[i++];
    p.option     = (const int*)d_in[i++];
    p.pos_score  = (const int*)d_in[i++];
    p.pos_option = (const int*)d_in[i++];
    p.neg_score  = (const int*)d_in[i++];
    p.neg_option = (const int*)d_in[i++];
  } else {
    // params concatenated into d_in[0] in insertion order
    const float* base = (const float*)d_in[0];
    size_t off = 0;
    auto take = [&](size_t n) { const float* r = base + off; off += n; return r; };
    p.qemb = take((size_t)(NQ_ + 1) * D_);
    p.remb = take((size_t)(NQ_ * NO_ + 2) * D_);
    for (int e = 0; e < 3; ++e) {
      p.ew1[e] = take(D_ * (D_ / 2)); p.eb1[e] = take(D_ / 2);
      p.ew2[e] = take((D_ / 2) * D_); p.eb2[e] = take(D_);
    }
    for (int a = 0; a < 3; ++a) {
      p.wq[a] = take(D_ * D_); p.bq[a] = take(D_);
      p.wv[a] = take(D_ * D_); p.bv[a] = take(D_);
      p.wo[a] = take(D_ * D_); p.bo[a] = take(D_);
      p.decay[a] = take(H_);
      p.lng[a] = take(D_); p.lnb[a] = take(D_);
    }
    p.question   = (const int*)d_in[1];
    p.score      = (const int*)d_in[3];
    p.option     = (const int*)d_in[4];
    p.pos_score  = (const int*)d_in[5];
    p.pos_option = (const int*)d_in[6];
    p.neg_score  = (const int*)d_in[7];
    p.neg_option = (const int*)d_in[8];
  }

  // ---- workspace carve-up ----
  char* cur = (char*)d_ws;
  auto alloc_f = [&](size_t n) { float* r = (float*)cur; cur += n * 4; return r; };
  auto alloc_h = [&](size_t n) { __bf16* r = (__bf16*)cur; cur += n * 2; return r; };
  float*  A   = alloc_f(BSD_);        // acc / dt / branch ht (f32)
  float*  Bb  = alloc_f(BSD_);        // ot f32 (for combine)
  float*  Cc  = alloc_f(BSD_);        // enc_correct out
  float*  Dv  = alloc_f(BSD_);        // enc_wrong out
  float*  F   = alloc_f(BSD_);        // dt_hat
  float*  G   = alloc_f(BSD_);        // qt_hat (persist)
  float*  Hh  = alloc_f(BSD_);        // enc_unchosen chunk out
  __bf16* Abf = alloc_h(BSD_);        // generic bf16 GEMM input
  __bf16* Qbf = alloc_h(BSD_);        // q (and k) bf16 row-major
  __bf16* VT  = alloc_h(BSD_);        // v bf16 [b,h,dk,s]
  __bf16* Ebf = alloc_h(BSD_);        // attention output bf16
  __bf16* H1  = alloc_h((size_t)BS_ * 128);
  __bf16 *wzq[3], *wzv[3], *wzo[3], *wz1[3], *wz2[3];
  for (int a = 0; a < 3; ++a) {
    wzq[a] = alloc_h(D_ * D_); wzv[a] = alloc_h(D_ * D_); wzo[a] = alloc_h(D_ * D_);
  }
  for (int e = 0; e < 3; ++e) { wz1[e] = alloc_h(D_ * 128); wz2[e] = alloc_h(128 * D_); }
  float* pa = alloc_f((size_t)B_ * D_);
  float* pb = alloc_f((size_t)B_ * D_);
  float* pc = alloc_f((size_t)B_ * D_);
  float* ht_out  = (float*)d_out;           // [B,S,D]
  float* cos_out = ht_out + BSD_;           // [B,2B]

  const dim3 ew_grid((unsigned)(BSD_ / 256));
  const dim3 ag(S_ / 16, H_, B_);

  auto swz = [&](const float* W, __bf16* out, int K, int N) {
    hipLaunchKernelGGL(swz_kernel, dim3((unsigned)((size_t)K * N / 256)), dim3(256),
                       0, stream, W, out, N);
  };
  // pre-swizzle all weights (cheap: ~1.5 MB total)
  for (int a = 0; a < 3; ++a) {
    swz(p.wq[a], wzq[a], D_, D_);
    swz(p.wv[a], wzv[a], D_, D_);
    swz(p.wo[a], wzo[a], D_, D_);
  }
  for (int e = 0; e < 3; ++e) {
    swz(p.ew1[e], wz1[e], D_, 128);
    swz(p.ew2[e], wz2[e], 128, D_);
  }

  auto gemm = [&](const __bf16* X, const __bf16* Wz, const float* bias, void* Y,
                  int N, int K, int mode) {
    dim3 g(BS_ / 16, N / 64);
    hipLaunchKernelGGL(gemm_kernel, g, dim3(32), 0, stream, X, Wz, bias, Y,
                       BS_, N, K, mode);
  };
  auto conv = [&](const float* x) {
    hipLaunchKernelGGL(conv_bf_kernel, ew_grid, dim3(256), 0, stream, x, Abf);
  };

  // preconv: Abf already holds the (shared) xq==xv input
  auto kt_layer = [&](const float* xq, const float* xv, int a, int maxout,
                      float* ydst, int preconv) {
    if (!preconv) conv(xq);
    gemm(Abf, wzq[a], p.bq[a], Qbf, D_, D_, OM_BF16);      // q (== k)
    if (!preconv && xv != xq) conv(xv);
    gemm(Abf, wzv[a], p.bv[a], VT, D_, D_, OM_VT);         // v, head-transposed
    hipLaunchKernelGGL(attn_kernel, ag, dim3(256), 0, stream, Qbf, VT, Ebf,
                       p.decay[a], maxout);
    gemm(Ebf, wzo[a], p.bo[a], ydst, D_, D_, OM_F32);      // out proj
    hipLaunchKernelGGL(ln_kernel, dim3(BS_), dim3(256), 0, stream, ydst,
                       p.lng[a], p.lnb[a]);
  };

  auto encode = [&](const int* score, const int* option) {  // result -> A
    hipLaunchKernelGGL(gather_ot_kernel, ew_grid, dim3(256), 0, stream,
                       p.question, option, p.remb, Bb, Abf);
    gemm(Abf, wz1[0], p.eb1[0], H1, 128, D_, OM_RELU_BF16);
    gemm(H1,  wz2[0], p.eb2[0], Cc, D_, 128, OM_F32);       // enc_correct(ot)
    gemm(Abf, wz1[1], p.eb1[1], H1, 128, D_, OM_RELU_BF16);
    gemm(H1,  wz2[1], p.eb2[1], Dv, D_, 128, OM_F32);       // enc_wrong(ot)
    for (int o = 0; o < NO_; ++o) {
      hipLaunchKernelGGL(gather_ut_kernel, ew_grid, dim3(256), 0, stream,
                         p.question, option, p.remb, Abf, o);
      gemm(Abf, wz1[2], p.eb1[2], H1, 128, D_, OM_RELU_BF16);
      gemm(H1,  wz2[2], p.eb2[2], Hh, D_, 128, OM_F32);     // enc_unchosen(ut_o)
      hipLaunchKernelGGL(accum_ut_kernel, ew_grid, dim3(256), 0, stream,
                         p.question, option, Hh, A, o, (o == 0) ? 1 : 0);
    }
    hipLaunchKernelGGL(combine_dt_kernel, ew_grid, dim3(256), 0, stream,
                       p.question, option, score, Bb, Cc, Dv, A);
  };

  // 1) qt gather (bf16 direct) + aq layer -> qt_hat (G)
  hipLaunchKernelGGL(gather_q_kernel, ew_grid, dim3(256), 0, stream,
                     p.question, p.qemb, Abf);
  kt_layer(nullptr, nullptr, /*aq*/1, /*maxout*/1, G, /*preconv*/1);

  // 2) main branch: encode -> ar -> as -> ht (d_out), pool -> pa
  encode(p.score, p.option);
  kt_layer(A, A, /*ar*/0, /*maxout*/0, F, 0);
  kt_layer(G, F, /*as*/2, /*maxout*/1, ht_out, 0);
  hipLaunchKernelGGL(pool_kernel, dim3(B_), dim3(256), 0, stream, ht_out, p.score, pa);

  // 3) positive branch
  encode(p.pos_score, p.pos_option);
  kt_layer(A, A, 0, 0, F, 0);
  kt_layer(G, F, 2, 1, A, 0);
  hipLaunchKernelGGL(pool_kernel, dim3(B_), dim3(256), 0, stream, A, p.score, pb);

  // 4) negative branch
  encode(p.neg_score, p.neg_option);
  kt_layer(A, A, 0, 0, F, 0);
  kt_layer(G, F, 2, 1, A, 0);
  hipLaunchKernelGGL(pool_kernel, dim3(B_), dim3(256), 0, stream, A, p.score, pc);

  // 5) inter cosine similarity [B, 2B]
  hipLaunchKernelGGL(cos_kernel, dim3(B_), dim3(64), 0, stream, pa, pb, pc, cos_out);
}
